// feature_fusion_64063732187319
// MI455X (gfx1250) — matrix operand order repaired and checked by reading, not compiled, not run on hardware
//
#include <hip/hip_runtime.h>
#include <hip/hip_bf16.h>

// ---- problem constants --------------------------------------------------
namespace {
constexpr int R_ = 4096;  // rays
constexpr int P_ = 128;   // points per ray
constexpr int D_ = 128;   // feature dim
constexpr int H_ = 256;   // hidden dim

// LDS layout (bytes) -- dynamic shared memory, one ray per workgroup
constexpr int OFF_DEP   = 0;        // bf16 [128][128]  depth
constexpr int OFF_RGB   = 32768;    // bf16 [128][128]  rgb
constexpr int OFF_RGBT  = 65536;    // bf16 [128][128]  rgb^T
constexpr int OFF_SEMT  = 98304;    // bf16 [128][128]  semantic^T   (aliased: x)
constexpr int OFF_SC    = 131072;   // f32  [128][128]  scores       (aliased: h bf16 [128][256])
constexpr int OFF_PR    = 196608;   // bf16 [128][128]  probabilities
constexpr int OFF_SF    = 229376;   // bf16 [128][128]  semantic_fused
constexpr int OFF_NR    = 262144;   // f32  [128]       rgb row sumsq
constexpr int OFF_NS    = 262656;   // f32  [128]       sf  row sumsq
constexpr int SMEM_BYTES = 263168;  // 257 KB  (< 320 KB WGP limit)
}

typedef __bf16 bf16t;
typedef __attribute__((ext_vector_type(16))) __bf16 v16bf;
typedef __attribute__((ext_vector_type(8)))  __bf16 v8bf;
typedef __attribute__((ext_vector_type(8)))  float  v8f;

#define WMMA_BF16(A, B, C) \
  __builtin_amdgcn_wmma_f32_16x16x32_bf16(false, (A), false, (B), (short)0, (C), false, false)

// A fragment (16x32 bf16, M x K) from a row-major [M][K] array, row stride ld.
// Lanes 0-15: row=lane, K=k0+0..7 (VGPR0-3), K=k0+16..23 (VGPR4-7).
// Lanes 16-31: row=lane-16, K=k0+8..15, K=k0+24..31.
__device__ __forceinline__ v16bf load_a_frag(const bf16t* base, int ld, int m0, int k0) {
  const int lane = threadIdx.x & 31;
  const bf16t* p = base + (m0 + (lane & 15)) * ld + k0 + ((lane >> 4) << 3);
  union { v16bf v; v8bf h[2]; } u;
  u.h[0] = *(const v8bf*)(p);
  u.h[1] = *(const v8bf*)(p + 16);
  return u.v;
}

// B fragment (32x16 bf16, K x N) where B[k][n] = M[n][k] and M is stored
// row-major [N][K] with row stride ld (i.e. operand supplied pre-transposed).
// Lanes 0-15: col=lane, K=k0+0..7, K=k0+16..23; lanes 16-31: K=k0+8..15, K=k0+24..31.
__device__ __forceinline__ v16bf load_b_frag(const bf16t* base, int ld, int n0, int k0) {
  const int lane = threadIdx.x & 31;
  const bf16t* p = base + (n0 + (lane & 15)) * ld + k0 + ((lane >> 4) << 3);
  union { v16bf v; v8bf h[2]; } u;
  u.h[0] = *(const v8bf*)(p);
  u.h[1] = *(const v8bf*)(p + 16);
  return u.v;
}

// Column softmax over axis i of S[i][j] (threads 0..127 each own one column j),
// writes normalized bf16 probabilities to Pm.
__device__ __forceinline__ void col_softmax(float* S, bf16t* Pm, int tid) {
  if (tid < P_) {
    const int j = tid;
    float mx = -3.402823466e38f;
    for (int i = 0; i < P_; ++i) mx = fmaxf(mx, S[i * P_ + j]);
    float sum = 0.f;
    for (int i = 0; i < P_; ++i) {
      float e = __expf(S[i * P_ + j] - mx);
      S[i * P_ + j] = e;
      sum += e;
    }
    const float inv = 1.f / sum;
    for (int i = 0; i < P_; ++i) Pm[i * P_ + j] = (bf16t)(S[i * P_ + j] * inv);
  }
}

// ---- weight pre-pass: f32 [in][out] -> bf16 transposed [out][in] --------
__global__ void prep_weights_kernel(const float* __restrict__ W1, const float* __restrict__ W2,
                                    bf16t* __restrict__ W1T, bf16t* __restrict__ W2T) {
  const int t = blockIdx.x * blockDim.x + threadIdx.x;
  if (t < 2 * D_ * H_) {                 // W1: [2D][H] -> W1T: [H][2D]
    const int k = t / H_, n = t % H_;
    W1T[n * (2 * D_) + k] = (bf16t)W1[t];
  } else {
    const int u = t - 2 * D_ * H_;
    if (u < H_ * D_) {                   // W2: [H][D] -> W2T: [D][H]
      const int k = u / D_, n = u % D_;
      W2T[n * H_ + k] = (bf16t)W2[u];
    }
  }
}

// ---- main fused kernel: one workgroup (256 thr = 8 waves) per ray -------
__global__ __launch_bounds__(256, 1)
void feature_fusion_kernel(const float* __restrict__ gSem, const float* __restrict__ gDep,
                           const float* __restrict__ gRgb,
                           const bf16t* __restrict__ gW1T, const float* __restrict__ gB1,
                           const bf16t* __restrict__ gW2T, const float* __restrict__ gB2,
                           float* __restrict__ gOutSf, float* __restrict__ gOutX) {
  extern __shared__ char smem[];
  bf16t* sDep  = (bf16t*)(smem + OFF_DEP);
  bf16t* sRgb  = (bf16t*)(smem + OFF_RGB);
  bf16t* sRgbT = (bf16t*)(smem + OFF_RGBT);
  bf16t* sSemT = (bf16t*)(smem + OFF_SEMT);
  float* sSc   = (float*)(smem + OFF_SC);
  bf16t* sPr   = (bf16t*)(smem + OFF_PR);
  bf16t* sSf   = (bf16t*)(smem + OFF_SF);
  float* sNr   = (float*)(smem + OFF_NR);
  float* sNs   = (float*)(smem + OFF_NS);
  bf16t* sX = sSemT;          // alias: semantic^T dead after phase 3
  bf16t* sH = (bf16t*)sSc;    // alias: f32 scores dead after phase 5

  const int tid  = threadIdx.x;
  const int lane = tid & 31;
  const int wave = tid >> 5;
  const int m0   = wave * 16;                       // each wave owns 16 output rows
  const int mb   = m0 + ((lane >> 4) << 3);         // first M row held by this lane's acc VGPRs
  const size_t rayOff = (size_t)blockIdx.x * (P_ * D_);

  const float* sem = gSem + rayOff;
  const float* dep = gDep + rayOff;
  const float* rgb = gRgb + rayOff;
  float* outSf = gOutSf + rayOff;
  float* outX  = gOutX + rayOff;

  if (tid < P_) { sNr[tid] = 0.f; sNs[tid] = 0.f; }
  __syncthreads();

  // ---- Phase 0: stage inputs as bf16, build transposes, rgb row sumsq ----
  for (int e4 = tid; e4 < P_ * D_ / 4; e4 += 256) {
    const int e = e4 << 2;
    const int r = e >> 7;          // row
    const int c = e & 127;         // col (multiple of 4)
    const float4 vd = ((const float4*)dep)[e4];
    sDep[r * D_ + c + 0] = (bf16t)vd.x; sDep[r * D_ + c + 1] = (bf16t)vd.y;
    sDep[r * D_ + c + 2] = (bf16t)vd.z; sDep[r * D_ + c + 3] = (bf16t)vd.w;
    const float4 vr = ((const float4*)rgb)[e4];
    sRgb[r * D_ + c + 0] = (bf16t)vr.x; sRgb[r * D_ + c + 1] = (bf16t)vr.y;
    sRgb[r * D_ + c + 2] = (bf16t)vr.z; sRgb[r * D_ + c + 3] = (bf16t)vr.w;
    sRgbT[(c + 0) * P_ + r] = (bf16t)vr.x; sRgbT[(c + 1) * P_ + r] = (bf16t)vr.y;
    sRgbT[(c + 2) * P_ + r] = (bf16t)vr.z; sRgbT[(c + 3) * P_ + r] = (bf16t)vr.w;
    atomicAdd(&sNr[r], vr.x * vr.x + vr.y * vr.y + vr.z * vr.z + vr.w * vr.w);
    const float4 vs = ((const float4*)sem)[e4];
    sSemT[(c + 0) * P_ + r] = (bf16t)vs.x; sSemT[(c + 1) * P_ + r] = (bf16t)vs.y;
    sSemT[(c + 2) * P_ + r] = (bf16t)vs.z; sSemT[(c + 3) * P_ + r] = (bf16t)vs.w;
  }
  __syncthreads();

  // ---- Phase 1: S1[i][j] = depth[i].rgb[j] / ||rgb[i]||  (f32 scores) ----
  {
    v16bf a[4];
#pragma unroll
    for (int k = 0; k < 4; ++k) a[k] = load_a_frag(sDep, D_, m0, k * 32);
    float rs[8];
#pragma unroll
    for (int v = 0; v < 8; ++v) rs[v] = rsqrtf(sNr[mb + v] + 1e-30f);
    for (int nt = 0; nt < 8; ++nt) {
      v8f acc = {0.f, 0.f, 0.f, 0.f, 0.f, 0.f, 0.f, 0.f};
#pragma unroll
      for (int k = 0; k < 4; ++k)
        acc = WMMA_BF16(a[k], load_b_frag(sRgb, D_, nt * 16, k * 32), acc);
      const int n = nt * 16 + (lane & 15);
#pragma unroll
      for (int v = 0; v < 8; ++v) sSc[(mb + v) * P_ + n] = acc[v] * rs[v];
    }
  }
  __syncthreads();

  // ---- Phase 2: softmax over axis i, write bf16 probs --------------------
  col_softmax(sSc, sPr, tid);
  __syncthreads();

  // ---- Phase 3: sf = P1 @ semantic ; emit bf16 + f32 out + row sumsq -----
  {
    v16bf a[4];
#pragma unroll
    for (int k = 0; k < 4; ++k) a[k] = load_a_frag(sPr, P_, m0, k * 32);
    float rowsq[8] = {0.f, 0.f, 0.f, 0.f, 0.f, 0.f, 0.f, 0.f};
    for (int nt = 0; nt < 8; ++nt) {
      v8f acc = {0.f, 0.f, 0.f, 0.f, 0.f, 0.f, 0.f, 0.f};
#pragma unroll
      for (int k = 0; k < 4; ++k)
        acc = WMMA_BF16(a[k], load_b_frag(sSemT, P_, nt * 16, k * 32), acc);
      const int n = nt * 16 + (lane & 15);
#pragma unroll
      for (int v = 0; v < 8; ++v) {
        const float f = acc[v];
        sSf[(mb + v) * D_ + n] = (bf16t)f;
        outSf[(mb + v) * D_ + n] = f;
        rowsq[v] += f * f;
      }
    }
#pragma unroll
    for (int v = 0; v < 8; ++v) atomicAdd(&sNs[mb + v], rowsq[v]);
  }
  __syncthreads();

  // ---- Phase 4: S2[i][j] = depth[i].sf[j] / ||sf[i]|| --------------------
  {
    v16bf a[4];
#pragma unroll
    for (int k = 0; k < 4; ++k) a[k] = load_a_frag(sDep, D_, m0, k * 32);
    float rs[8];
#pragma unroll
    for (int v = 0; v < 8; ++v) rs[v] = rsqrtf(sNs[mb + v] + 1e-30f);
    for (int nt = 0; nt < 8; ++nt) {
      v8f acc = {0.f, 0.f, 0.f, 0.f, 0.f, 0.f, 0.f, 0.f};
#pragma unroll
      for (int k = 0; k < 4; ++k)
        acc = WMMA_BF16(a[k], load_b_frag(sSf, D_, nt * 16, k * 32), acc);
      const int n = nt * 16 + (lane & 15);
#pragma unroll
      for (int v = 0; v < 8; ++v) sSc[(mb + v) * P_ + n] = acc[v] * rs[v];
    }
  }
  __syncthreads();

  // ---- Phase 5: second softmax ------------------------------------------
  col_softmax(sSc, sPr, tid);
  __syncthreads();

  // ---- Phase 6: x = P2 @ rgb  (bf16, into sX = old semantic^T) -----------
  {
    v16bf a[4];
#pragma unroll
    for (int k = 0; k < 4; ++k) a[k] = load_a_frag(sPr, P_, m0, k * 32);
    for (int nt = 0; nt < 8; ++nt) {
      v8f acc = {0.f, 0.f, 0.f, 0.f, 0.f, 0.f, 0.f, 0.f};
#pragma unroll
      for (int k = 0; k < 4; ++k)
        acc = WMMA_BF16(a[k], load_b_frag(sRgbT, P_, nt * 16, k * 32), acc);
      const int n = nt * 16 + (lane & 15);
#pragma unroll
      for (int v = 0; v < 8; ++v) sX[(mb + v) * D_ + n] = (bf16t)acc[v];
    }
  }
  __syncthreads();

  // ---- Phase 7: h = relu([x, rgb] @ W1 + b1)  (128x256, into sH) ---------
  {
    v16bf a[8];
#pragma unroll
    for (int k = 0; k < 4; ++k) a[k] = load_a_frag(sX, D_, m0, k * 32);
#pragma unroll
    for (int k = 4; k < 8; ++k) a[k] = load_a_frag(sRgb, D_, m0, k * 32 - 128);
    for (int nt = 0; nt < 16; ++nt) {
      v8f acc = {0.f, 0.f, 0.f, 0.f, 0.f, 0.f, 0.f, 0.f};
#pragma unroll
      for (int k = 0; k < 8; ++k)
        acc = WMMA_BF16(a[k], load_b_frag(gW1T, 2 * D_, nt * 16, k * 32), acc);
      const int n = nt * 16 + (lane & 15);
      const float bias = gB1[n];
#pragma unroll
      for (int v = 0; v < 8; ++v)
        sH[(mb + v) * H_ + n] = (bf16t)fmaxf(acc[v] + bias, 0.f);
    }
  }
  __syncthreads();

  // ---- Phase 8: out = h @ W2 + b2  (f32 to global) -----------------------
  {
    v16bf a[8];
#pragma unroll
    for (int k = 0; k < 8; ++k) a[k] = load_a_frag(sH, H_, m0, k * 32);
    for (int nt = 0; nt < 8; ++nt) {
      v8f acc = {0.f, 0.f, 0.f, 0.f, 0.f, 0.f, 0.f, 0.f};
#pragma unroll
      for (int k = 0; k < 8; ++k)
        acc = WMMA_BF16(a[k], load_b_frag(gW2T, H_, nt * 16, k * 32), acc);
      const int n = nt * 16 + (lane & 15);
      const float bias = gB2[n];
#pragma unroll
      for (int v = 0; v < 8; ++v)
        outX[(mb + v) * D_ + n] = acc[v] + bias;
    }
  }
}

// ---- host-side launch ---------------------------------------------------
extern "C" void kernel_launch(void* const* d_in, const int* in_sizes, int n_in,
                              void* d_out, int out_size, void* d_ws, size_t ws_size,
                              hipStream_t stream) {
  const float* sem = (const float*)d_in[0];
  const float* dep = (const float*)d_in[1];
  const float* rgb = (const float*)d_in[2];
  const float* W1  = (const float*)d_in[3];
  const float* b1  = (const float*)d_in[4];
  const float* W2  = (const float*)d_in[5];
  const float* b2  = (const float*)d_in[6];
  float* out = (float*)d_out;

  bf16t* W1T = (bf16t*)d_ws;                 // [H][2D] bf16 : 128 KB
  bf16t* W2T = W1T + (size_t)2 * D_ * H_;    // [D][H]  bf16 :  64 KB

  const int prepThreads = 2 * D_ * H_ + H_ * D_;   // 98304
  prep_weights_kernel<<<(prepThreads + 255) / 256, 256, 0, stream>>>(W1, W2, W1T, W2T);

  feature_fusion_kernel<<<R_, 256, SMEM_BYTES, stream>>>(
      sem, dep, rgb, W1T, b1, W2T, b2,
      out, out + (size_t)R_ * P_ * D_);
}
